// YinPitch_37452114821248
// MI455X (gfx1250) — matrix operand order repaired and checked by reading, not compiled, not run on hardware
//
#include <hip/hip_runtime.h>

typedef __attribute__((ext_vector_type(16))) _Float16 v16h;
typedef __attribute__((ext_vector_type(8)))  float    v8f;

typedef unsigned int u32;
typedef u32 u32x4 __attribute__((ext_vector_type(4)));
typedef int i32x4 __attribute__((ext_vector_type(4)));
typedef int i32x8 __attribute__((ext_vector_type(8)));

#define BATCH     64
#define TLEN      160000
#define STRIDE_S  160
#define NFRAMES   1001      // TLEN/STRIDE_S + 1
#define WLEN      257       // frame window
#define TAUMAX    213       // 16000 // 75
#define TAUMIN    26        // 16000 // 600
#define CLEN      187       // TAUMAX - TAUMIN
#define NTG       14        // lag groups of 16  -> covers tau 0..223
#define NKC       9         // K chunks of 32    -> covers j 0..287
#define FPAD      512       // padded f16 frame (max shifted read = 510)
#define THRESH    0.1f

// All LDS in one struct so the TDM target (f32 frame) sits at static LDS
// offset 0 -> D#.lds_addr is a plain constant, no addrspace casts needed.
struct Smem {
    float    f32[WLEN];      // TDM destination (LDS offset 0)
    _Float16 f16[FPAD];      // f16 frame, zero-padded for shifted reads
    float    cs [WLEN];      // cumsum of squares
    float    corr[NTG * 16]; // corr[0..223]
    float    cd [TAUMAX];    // diff, then cd (in place)
    float    ps [TAUMAX];    // prefix sum of diff[1..t]
};

// One wave (32 lanes) per frame. Autocorrelation via broadcast-row WMMA:
//   A(16x32): every row = frame[k0 .. k0+31]            (identical rows)
//   B(32x16): column n  = frame[k0+K + t0 + n]
//   => every row of D accumulates corr[t0+n]; exact full-lag sums after
//      accumulating the 9 K-chunks (frame zero-padded beyond sample 256).
__global__ __launch_bounds__(32)
void yin_frame_kernel(const float* __restrict__ in, float* __restrict__ pitchBuf)
{
    __shared__ Smem sm;

    const int lane = threadIdx.x;        // wave32
    const int fb   = blockIdx.x;         // b*NFRAMES + f
    const int b    = fb / NFRAMES;
    const int f    = fb - b * NFRAMES;
    const long sig = (long)b * TLEN;
    const long fbase = (long)f * STRIDE_S;

    // ---- stage f32 frame into LDS via the Tensor Data Mover ----
    // 1-row tile of WLEN f32; tensor_dim0 = remaining samples so the TDM's
    // OOB-read-returns-zero implements the reference's end-of-signal padding.
#if __has_builtin(__builtin_amdgcn_tensor_load_to_lds) && __has_builtin(__builtin_amdgcn_s_wait_tensorcnt)
    {
        const unsigned long long ga =
            (unsigned long long)(const void*)(in + sig + fbase);
        long remL = (long)TLEN - fbase;
        const u32 dim0 = (u32)(remL > 0 ? remL : 0);

        u32x4 g0;
        g0.x = 1u;                                   // count=1 (valid D#)
        g0.y = 0u;                                   // lds_addr = offset of sm.f32
        g0.z = (u32)(ga & 0xFFFFFFFFull);            // global_addr[31:0]
        g0.w = (u32)((ga >> 32) & 0x01FFFFFFull)     // global_addr[56:32]
             | (2u << 30);                           // type=2 ("image")

        i32x8 g1;
        g1[0] = (int)(2u << 16);                     // data_size=2 -> 4 bytes
        g1[1] = (int)((dim0 & 0xFFFFu) << 16);       // tensor_dim0[15:0]
        g1[2] = (int)(((dim0 >> 16) & 0xFFFFu)       // tensor_dim0[31:16]
             | (1u << 16));                          // tensor_dim1 = 1
        g1[3] = (int)((u32)WLEN << 16);              // tile_dim0 = 257
        g1[4] = 1;                                   // tile_dim1 = 1
        g1[5] = (int)dim0;                           // tensor_dim0_stride (unused, 1 row)
        g1[6] = 0;
        g1[7] = 0;

        i32x4 gz = {0, 0, 0, 0};                     // groups 2/3: <=2D tensor
#if defined(__clang_major__) && (__clang_major__ >= 23)
        i32x8 gz8 = {0, 0, 0, 0, 0, 0, 0, 0};
        __builtin_amdgcn_tensor_load_to_lds(g0, g1, gz, gz, gz8, 0);
#else
        __builtin_amdgcn_tensor_load_to_lds(g0, g1, gz, gz, 0);
#endif
        __builtin_amdgcn_s_wait_tensorcnt(0);
    }
#else
    // Fallback: manual staging through VGPRs.
    if (lane == 0) __builtin_prefetch(in + sig + fbase, 0, 3);
    for (int j = lane; j < WLEN; j += 32) {
        const long gi = fbase + j;
        sm.f32[j] = (gi < TLEN) ? in[sig + gi] : 0.0f;
    }
#endif
    __syncthreads();

    // ---- convert to zero-padded f16 frame for the matrix pipe ----
    for (int j = lane; j < FPAD; j += 32)
        sm.f16[j] = (j < WLEN) ? (_Float16)sm.f32[j] : (_Float16)0.0f;
    __syncthreads();

    // ---- lag correlation on the matrix pipe ----
    const int nIdx = lane & 15;
    const int hi   = (lane >> 4) & 1;    // lane half selects K sub-block
    for (int g = 0; g < NTG; ++g) {
        const int t0 = g << 4;
        v8f acc = {};
        for (int c = 0; c < NKC; ++c) {
            const int k0 = c << 5;
            v16h a, bm;
            // A 16x32 f16 layout: lanes 0-15 hold K=0..15,
            //                     lanes 16-31 hold K=16..31 (matches B).
            const int abase = k0 + hi * 16;
            #pragma unroll
            for (int i = 0; i < 16; ++i) {
                a[i]     = sm.f16[abase + i];
            }
            // B 32x16 f16 layout: lanes 0-15 hold K=0..15 (col N=lane),
            //                     lanes 16-31 hold K=16..31 (col N=lane-16).
            const int bbase = k0 + hi * 16 + t0 + nIdx;
            #pragma unroll
            for (int i = 0; i < 16; ++i) bm[i] = sm.f16[bbase + i];
            acc = __builtin_amdgcn_wmma_f32_16x16x32_f16(
                      false, a, false, bm, (short)0, acc, false, false);
        }
        // D VGPR0, lanes 0-15 = row M=0 -> corr[t0 + n]
        if (lane < 16) sm.corr[t0 + lane] = acc[0];
    }
    __syncthreads();

    // ---- energy cumsum (tiny serial scan) ----
    if (lane == 0) {
        float run = 0.0f;
        for (int j = 0; j < WLEN; ++j) { const float x = sm.f32[j]; run += x * x; sm.cs[j] = run; }
    }
    __syncthreads();

    // ---- YIN difference:  d[t] = cs[256-t] - 2 corr[t] + cs[256] - cs[t] ----
    for (int t = lane; t < TAUMAX; t += 32)
        sm.cd[t] = sm.cs[WLEN - 1 - t] - 2.0f * sm.corr[t] + sm.cs[WLEN - 1] - sm.cs[t];
    __syncthreads();

    if (lane == 0) {                     // running sum of d[1..t]
        float run = 0.0f; sm.ps[0] = 0.0f;
        for (int t = 1; t < TAUMAX; ++t) { run += sm.cd[t]; sm.ps[t] = run; }
    }
    __syncthreads();

    // ---- cumulative-mean normalization (in place) ----
    for (int t = lane; t < TAUMAX; t += 32) {
        const float cdv = (t == 0) ? 1.0f : sm.cd[t] * (float)t / (sm.ps[t] + 1e-7f);
        sm.cd[t] = cdv;
    }
    __syncthreads();

    // ---- threshold / local-min / parabolic refine -> pitch ----
    if (lane == 0) {
        const float* cmnd = &sm.cd[TAUMIN];          // length CLEN
        int thold = 0;
        for (int i = 0; i < CLEN; ++i) { if (cmnd[i] < THRESH) { thold = i; break; } }
        if (thold == 0) thold = CLEN;                // reference semantics
        int tau = 0;
        for (int i = thold; i < CLEN; ++i) {
            const float dl = (i == 0)        ? 0.0f : cmnd[i]     - cmnd[i - 1];
            const float dr = (i == CLEN - 1) ? 0.0f : cmnd[i + 1] - cmnd[i];
            if (dl <= 0.0f && dr >= 0.0f) { tau = i; break; }
        }
        float shift = 0.0f;
        if (tau >= 1 && tau <= CLEN - 2) {
            const float p  = cmnd[tau - 1], cc = cmnd[tau], nn = cmnd[tau + 1];
            const float aq = nn + p - 2.0f * cc;
            const float bq = 0.5f * (nn - p);
            shift = (fabsf(bq) >= fabsf(aq)) ? 0.0f : (-bq / aq);
        }
        float pitch = 0.0f;
        if (tau > 0) pitch = 16000.0f / ((float)(tau + TAUMIN + 1) + shift);
        pitchBuf[fb] = pitch;
    }
}

// 30-tap running median over frames with edge padding (15 left, 14 right).
__global__ __launch_bounds__(256)
void yin_median_kernel(const float* __restrict__ pitchBuf, float* __restrict__ out)
{
    const int idx = blockIdx.x * blockDim.x + threadIdx.x;
    if (idx >= BATCH * NFRAMES) return;
    const int b = idx / NFRAMES;
    const int f = idx - b * NFRAMES;

    float w[30];
    #pragma unroll
    for (int j = 0; j < 30; ++j) {
        int k = f + j - 15;                          // edge ("clamp") padding
        k = k < 0 ? 0 : (k > NFRAMES - 1 ? NFRAMES - 1 : k);
        w[j] = pitchBuf[b * NFRAMES + k];
    }
    #pragma unroll
    for (int i = 1; i < 30; ++i) {                   // insertion sort
        const float key = w[i];
        int j = i - 1;
        while (j >= 0 && w[j] > key) { w[j + 1] = w[j]; --j; }
        w[j + 1] = key;
    }
    out[idx] = w[14];                                // (30-1)//2
}

extern "C" void kernel_launch(void* const* d_in, const int* in_sizes, int n_in,
                              void* d_out, int out_size, void* d_ws, size_t ws_size,
                              hipStream_t stream)
{
    (void)in_sizes; (void)n_in; (void)out_size; (void)ws_size;
    const float* in       = (const float*)d_in[0];
    float*       out      = (float*)d_out;
    float*       pitchBuf = (float*)d_ws;            // 64*1001 f32 = 256 KB

    yin_frame_kernel<<<BATCH * NFRAMES, 32, 0, stream>>>(in, pitchBuf);
    yin_median_kernel<<<(BATCH * NFRAMES + 255) / 256, 256, 0, stream>>>(pitchBuf, out);
}
